// MeteoGraphSAGE_2954937500043
// MI455X (gfx1250) — matrix operand order repaired and checked by reading, not compiled, not run on hardware
//
#include <hip/hip_runtime.h>
#include <hip/hip_bf16.h>

typedef __attribute__((ext_vector_type(16))) __bf16 v16bf;
typedef __attribute__((ext_vector_type(8)))  float  v8f;
typedef int v4i __attribute__((ext_vector_type(4)));
typedef int v8i __attribute__((ext_vector_type(8)));

#define kT 8
#define kF 64
#define kH 256
#define kO 8
#define kL 2
#define kEps 1e-5f

__device__ __forceinline__ float sigf(float x) { return 1.0f / (1.0f + __expf(-x)); }
__device__ __forceinline__ float tanhf_(float x) {
  float e = __expf(2.0f * x);
  return (e - 1.0f) / (e + 1.0f);
}

// ---------------- utility / elementwise kernels ----------------

__global__ void k_fill(float* __restrict__ p, float v, long long n) {
  long long i = blockIdx.x * (long long)blockDim.x + threadIdx.x;
  if (i < n) p[i] = v;
}

__global__ void k_addvec(const float* __restrict__ a, const float* __restrict__ b,
                         float* __restrict__ o, int n) {
  int i = blockIdx.x * blockDim.x + threadIdx.x;
  if (i < n) o[i] = a[i] + b[i];
}

__global__ void k_count(const int* __restrict__ dst, float* __restrict__ cnt, int E) {
  int i = blockIdx.x * blockDim.x + threadIdx.x;
  if (i < E) atomicAdd(&cnt[dst[i]], 1.0f);
}

__global__ void k_invdenom(float* __restrict__ c, int n) {
  int i = blockIdx.x * blockDim.x + threadIdx.x;
  if (i < n) c[i] = 1.0f / fmaxf(c[i], 1.0f);
}

// agg[dst] += h[src] * inv_denom[dst]; one thread per (edge, channel)
__global__ void k_scatter(const float* __restrict__ h, const int* __restrict__ src,
                          const int* __restrict__ dst, const float* __restrict__ invd,
                          float* __restrict__ agg, long long total) {
  long long i = blockIdx.x * (long long)blockDim.x + threadIdx.x;
  if (i >= total) return;
  int c = (int)(i & (kH - 1));
  int e = (int)(i >> 8);
  int s = src[e];
  int d = dst[e];
  atomicAdd(&agg[(size_t)d * kH + c], h[(size_t)s * kH + c] * invd[d]);
}

// per-column sum / sumsq over the node dim; one thread per column per block
__global__ void k_bnstats(const float* __restrict__ comb, float* __restrict__ stats, int N) {
  int c = threadIdx.x;  // kH threads
  float s = 0.f, ss = 0.f;
  for (int r = blockIdx.x; r < N; r += gridDim.x) {
    float v = comb[(size_t)r * kH + c];
    s += v;
    ss += v * v;
  }
  atomicAdd(&stats[c], s);
  atomicAdd(&stats[kH + c], ss);
}

// h += relu( (comb-mu)*rsqrt(var+eps)*gamma + beta )
__global__ void k_bnapply(const float* __restrict__ comb, const float* __restrict__ stats,
                          const float* __restrict__ gamma, const float* __restrict__ beta,
                          float* __restrict__ h, long long total, float invN) {
  long long i = blockIdx.x * (long long)blockDim.x + threadIdx.x;
  if (i >= total) return;
  int c = (int)(i & (kH - 1));
  float mu = stats[c] * invN;
  float var = stats[kH + c] * invN - mu * mu;
  float v = (comb[i] - mu) * rsqrtf(var + kEps) * gamma[c] + beta[c];
  h[i] += fmaxf(v, 0.0f);
}

// gate order i,f,g,o (PyTorch); gates layout [N, 4H]
__global__ void k_lstm(const float* __restrict__ gates, float* __restrict__ hS,
                       float* __restrict__ cS, long long total) {
  long long i = blockIdx.x * (long long)blockDim.x + threadIdx.x;
  if (i >= total) return;
  int c = (int)(i & (kH - 1));
  long long n = i >> 8;
  const float* g = gates + n * (4 * kH);
  float ig = sigf(g[c]);
  float fg = sigf(g[kH + c]);
  float gg = tanhf_(g[2 * kH + c]);
  float og = sigf(g[3 * kH + c]);
  float cn = fg * cS[i] + ig * gg;
  cS[i] = cn;
  hS[i] = og * tanhf_(cn);
}

// out[N,8] = h_last @ W_dec + b_dec (tiny; plain VALU)
__global__ void k_decoder(const float* __restrict__ h, const float* __restrict__ W,
                          const float* __restrict__ b, float* __restrict__ out, int N) {
  int i = blockIdx.x * blockDim.x + threadIdx.x;
  if (i >= N * kO) return;
  int n = i >> 3, o = i & 7;
  float acc = b[o];
  const float* hr = h + (size_t)n * kH;
#pragma unroll 8
  for (int k = 0; k < kH; ++k) acc = fmaf(hr[k], W[k * kO + o], acc);
  out[i] = acc;
}

// ---------------- WMMA GEMM ----------------
// Pack fp32 weights [K,Ncols] (or [Ncols,K] if transposed) into bf16 B-fragment
// order: element e of lane l in fragment (kt,nt) holds B[kt*32 + (l<16?0:8) + (e&7) + (e<8?0:16), nt*16 + (l&15)]
__global__ void k_packb(const float* __restrict__ W, __bf16* __restrict__ Bp,
                        int K, int Ncols, int transposed) {
  int tid = blockIdx.x * blockDim.x + threadIdx.x;
  if (tid >= K * Ncols) return;
  int e = tid & 15;
  int lane = (tid >> 4) & 31;
  int frag = tid >> 9;
  int ntiles = Ncols >> 4;
  int nt = frag % ntiles;
  int kt = frag / ntiles;
  int k = kt * 32 + ((lane & 16) ? 8 : 0) + (e & 7) + ((e & 8) ? 16 : 0);
  int n = nt * 16 + (lane & 15);
  float v = transposed ? W[(size_t)n * K + k] : W[(size_t)k * Ncols + n];
  Bp[tid] = (__bf16)v;
}

// Raw fp32 A-tile registers for software pipelining (converted to bf16 at use).
struct AF { float4 a, b, c, d; };

__device__ __forceinline__ AF load_a_raw(const float* __restrict__ A, int row0, int lda,
                                         int kk, int lane) {
  int m = row0 + (lane & 15);
  int kb = (lane & 16) ? 8 : 0;
  const float* p = A + (size_t)m * lda + kk + kb;
  AF r;
  r.a = *(const float4*)(p);
  r.b = *(const float4*)(p + 4);
  r.c = *(const float4*)(p + 16);
  r.d = *(const float4*)(p + 20);
  return r;
}

__device__ __forceinline__ v16bf cvt_a(const AF& f) {
  v16bf a;
  a[0] = (__bf16)f.a.x;  a[1] = (__bf16)f.a.y;  a[2] = (__bf16)f.a.z;  a[3] = (__bf16)f.a.w;
  a[4] = (__bf16)f.b.x;  a[5] = (__bf16)f.b.y;  a[6] = (__bf16)f.b.z;  a[7] = (__bf16)f.b.w;
  a[8] = (__bf16)f.c.x;  a[9] = (__bf16)f.c.y;  a[10] = (__bf16)f.c.z; a[11] = (__bf16)f.c.w;
  a[12] = (__bf16)f.d.x; a[13] = (__bf16)f.d.y; a[14] = (__bf16)f.d.z; a[15] = (__bf16)f.d.w;
  return a;
}

__device__ __forceinline__ v16bf load_b_lds(const __bf16* __restrict__ p) {
  v16bf b;
#pragma unroll
  for (int e = 0; e < 16; ++e) b[e] = p[e];
  return b;
}

// Tensor Data Mover: DMA a 2-D strip of packed B fragments into LDS.
// 2-group D# (2-D tensor): data_size=8B, tile = rowBytes8 x rows,
// row stride (global) = strideBytes8 (8-byte units); LDS fills contiguously.
__device__ __forceinline__ void tdm_load_b(const __bf16* gsrc, unsigned ldsByteAddr,
                                           unsigned rowBytes8, unsigned strideBytes8,
                                           unsigned rows) {
  unsigned long long ga = (unsigned long long)(size_t)gsrc;
  v4i g0;
  g0[0] = 1;                                              // count=1, user mode
  g0[1] = (int)ldsByteAddr;                               // lds_addr (bytes)
  g0[2] = (int)(unsigned)(ga & 0xffffffffull);            // global_addr[31:0]
  g0[3] = (int)((unsigned)((ga >> 32) & 0x1ffffffull) | (2u << 30));  // [56:32] | type=2
  v8i g1;
  g1[0] = (int)(3u << 16);                                // wg_mask=0 (not in cluster), data_size=8B
  g1[1] = (int)((strideBytes8 & 0xffffu) << 16);          // tensor_dim0[15:0]
  g1[2] = (int)(((strideBytes8 >> 16) & 0xffffu) | (rows << 16));  // dim0[31:16] | tensor_dim1[15:0]
  g1[3] = (int)(rowBytes8 << 16);                         // tensor_dim1[31:16]=0 | tile_dim0
  g1[4] = (int)rows;                                      // tile_dim1 | tile_dim2=0
  g1[5] = (int)strideBytes8;                              // tensor_dim0_stride[31:0]
  g1[6] = 0;                                              // stride[47:32] | dim1_stride[15:0]=0
  g1[7] = 0;
  asm volatile("tensor_load_to_lds %0, %1" :: "s"(g0), "s"(g1) : "memory");
}

// C = A1@B1 (+ A2@B2) + bias ; 4 waves/block, each wave a 16x64 strip (64x64 tile).
// B strip staged once per block into LDS by the TDM; A double-buffered in VGPRs.
__global__ void k_gemm(const float* __restrict__ A1, const __bf16* __restrict__ B1,
                       const float* __restrict__ A2, const __bf16* __restrict__ B2,
                       const float* __restrict__ bias, float* __restrict__ C,
                       int M, int K, int Ncols) {
  __shared__ __bf16 ldsB[16 * 2048];  // up to 8 K-chunks x (single|dual) x 4KB = 64KB
  int lane = threadIdx.x & 31;
  int wave = threadIdx.x >> 5;
  int row0 = blockIdx.x * 64 + wave * 16;
  int ntiles = Ncols >> 4;
  int ntBase = blockIdx.y * 4;
  int kts = K >> 5;
  bool active = (row0 + 16 <= M);  // wave-uniform (M multiple of 16)

  if (wave == 0) {  // one TDM issue per block; EXEC-independent DMA
    unsigned ldsBase = (unsigned)(size_t)&ldsB[0];
    tdm_load_b(B1 + (size_t)ntBase * 512, ldsBase, 512u, (unsigned)(ntiles * 128), (unsigned)kts);
    if (A2)
      tdm_load_b(B2 + (size_t)ntBase * 512, ldsBase + (unsigned)kts * 4096u, 512u,
                 (unsigned)(ntiles * 128), (unsigned)kts);
    __builtin_amdgcn_s_wait_tensorcnt(0);
  }
  __syncthreads();
  if (!active) return;

  v8f acc[4];
#pragma unroll
  for (int s = 0; s < 4; ++s) {
    float bv = bias[(ntBase + s) * 16 + (lane & 15)];
#pragma unroll
    for (int r = 0; r < 8; ++r) acc[s][r] = bv;
  }

  AF fa1 = load_a_raw(A1, row0, K, 0, lane);
  AF fa2{};
  if (A2) fa2 = load_a_raw(A2, row0, K, 0, lane);

  for (int kt = 0; kt < kts; ++kt) {
    AF na1 = fa1, na2 = fa2;
    if (kt + 1 < kts) {  // prefetch next A chunk behind this iteration's WMMAs
      na1 = load_a_raw(A1, row0, K, (kt + 1) * 32, lane);
      if (A2) na2 = load_a_raw(A2, row0, K, (kt + 1) * 32, lane);
    }
    v16bf a1 = cvt_a(fa1);
#pragma unroll
    for (int s = 0; s < 4; ++s) {
      v16bf b = load_b_lds(&ldsB[kt * 2048 + s * 512 + lane * 16]);
      acc[s] = __builtin_amdgcn_wmma_f32_16x16x32_bf16(false, a1, false, b, (short)0,
                                                       acc[s], false, false);
    }
    if (A2) {
      v16bf a2 = cvt_a(fa2);
#pragma unroll
      for (int s = 0; s < 4; ++s) {
        v16bf b = load_b_lds(&ldsB[(kts + kt) * 2048 + s * 512 + lane * 16]);
        acc[s] = __builtin_amdgcn_wmma_f32_16x16x32_bf16(false, a2, false, b, (short)0,
                                                         acc[s], false, false);
      }
    }
    fa1 = na1;
    fa2 = na2;
  }

  // C/D layout: element r -> row row0 + r + (lane<16?0:8), col = lane&15
  int m0 = row0 + ((lane & 16) ? 8 : 0);
  int nn = lane & 15;
#pragma unroll
  for (int s = 0; s < 4; ++s) {
    float* cp = C + (size_t)m0 * Ncols + (ntBase + s) * 16 + nn;
#pragma unroll
    for (int r = 0; r < 8; ++r) cp[(size_t)r * Ncols] = acc[s][r];
  }
}

// ---------------- host orchestration ----------------

extern "C" void kernel_launch(void* const* d_in, const int* in_sizes, int n_in,
                              void* d_out, int out_size, void* d_ws, size_t ws_size,
                              hipStream_t stream) {
  (void)n_in; (void)out_size; (void)ws_size;
  const float* x       = (const float*)d_in[0];
  const float* W0      = (const float*)d_in[1];
  const float* b0      = (const float*)d_in[2];
  const float* Ws_self = (const float*)d_in[3];
  const float* bs_self = (const float*)d_in[4];
  const float* Ws_nei  = (const float*)d_in[5];
  const float* bs_nei  = (const float*)d_in[6];
  const float* gamma   = (const float*)d_in[7];
  const float* beta    = (const float*)d_in[8];
  const float* W_ih    = (const float*)d_in[9];
  const float* W_hh    = (const float*)d_in[10];
  const float* b_ih    = (const float*)d_in[11];
  const float* b_hh    = (const float*)d_in[12];
  const float* W_dec   = (const float*)d_in[13];
  const float* b_dec   = (const float*)d_in[14];
  const int*   eidx    = (const int*)d_in[15];

  const int N = in_sizes[0] / (kT * kF);  // 50000 (multiple of 16)
  const int E = in_sizes[15] / 2;
  const int* src = eidx;
  const int* dst = eidx + E;

  char* wsp = (char*)d_ws;
  auto take = [&](size_t bytes) {
    char* p = wsp;
    wsp += (bytes + 255) & ~(size_t)255;
    return p;
  };
  float*  h     = (float*)take((size_t)N * kH * 4);
  float*  agg   = (float*)take((size_t)N * kH * 4);
  float*  comb  = (float*)take((size_t)N * kH * 4);
  float*  gates = (float*)take((size_t)N * 4 * kH * 4);
  float*  hS    = (float*)take((size_t)N * kH * 4);
  float*  cS    = (float*)take((size_t)N * kH * 4);
  float*  invd  = (float*)take((size_t)N * 4);
  float*  stats = (float*)take(2 * kH * 4);
  float*  biasg = (float*)take(4 * kH * 4);
  float*  bsum  = (float*)take(kL * kH * 4);
  __bf16* W0p   = (__bf16*)take((size_t)kF * kH * 2);
  __bf16* Wsp   = (__bf16*)take((size_t)kL * kH * kH * 2);
  __bf16* Wnp   = (__bf16*)take((size_t)kL * kH * kH * 2);
  __bf16* Wip   = (__bf16*)take((size_t)kH * 4 * kH * 2);
  __bf16* Whp   = (__bf16*)take((size_t)kH * 4 * kH * 2);

  auto cdiv = [](long long a, long long b) { return (int)((a + b - 1) / b); };

  // --- weight packing (bf16 B-fragment order) + fused biases ---
  k_packb<<<cdiv(kF * kH, 256), 256, 0, stream>>>(W0, W0p, kF, kH, 0);
  for (int i = 0; i < kL; ++i) {
    k_packb<<<cdiv(kH * kH, 256), 256, 0, stream>>>(Ws_self + (size_t)i * kH * kH,
                                                    Wsp + (size_t)i * kH * kH, kH, kH, 0);
    k_packb<<<cdiv(kH * kH, 256), 256, 0, stream>>>(Ws_nei + (size_t)i * kH * kH,
                                                    Wnp + (size_t)i * kH * kH, kH, kH, 0);
    k_addvec<<<1, kH, 0, stream>>>(bs_self + i * kH, bs_nei + i * kH, bsum + i * kH, kH);
  }
  k_packb<<<cdiv(kH * 4 * kH, 256), 256, 0, stream>>>(W_ih, Wip, kH, 4 * kH, 1);  // W_ih.T
  k_packb<<<cdiv(kH * 4 * kH, 256), 256, 0, stream>>>(W_hh, Whp, kH, 4 * kH, 1);  // W_hh.T
  k_addvec<<<cdiv(4 * kH, 256), 256, 0, stream>>>(b_ih, b_hh, biasg, 4 * kH);

  // --- degrees -> inverse denom ---
  k_fill<<<cdiv(N, 256), 256, 0, stream>>>(invd, 0.f, N);
  k_count<<<cdiv(E, 256), 256, 0, stream>>>(dst, invd, E);
  k_invdenom<<<cdiv(N, 256), 256, 0, stream>>>(invd, N);

  // --- LSTM state ---
  long long NH = (long long)N * kH;
  k_fill<<<cdiv(NH, 256), 256, 0, stream>>>(hS, 0.f, NH);
  k_fill<<<cdiv(NH, 256), 256, 0, stream>>>(cS, 0.f, NH);

  dim3 blk(128);  // 4 wave32 per block
  for (int t = 0; t < kT; ++t) {
    const float* xt = x + (size_t)t * N * kF;
    // h = x_t @ W0 + b0
    k_gemm<<<dim3(cdiv(N, 64), kH / 64), blk, 0, stream>>>(xt, W0p, nullptr, nullptr, b0,
                                                           h, N, kF, kH);
    for (int i = 0; i < kL; ++i) {
      k_fill<<<cdiv(NH, 256), 256, 0, stream>>>(agg, 0.f, NH);
      k_scatter<<<cdiv((long long)E * kH, 256), 256, 0, stream>>>(h, src, dst, invd, agg,
                                                                  (long long)E * kH);
      // comb = h@Ws_self + agg@Ws_nei + (bs_self+bs_nei)
      k_gemm<<<dim3(cdiv(N, 64), kH / 64), blk, 0, stream>>>(
          h, Wsp + (size_t)i * kH * kH, agg, Wnp + (size_t)i * kH * kH, bsum + i * kH,
          comb, N, kH, kH);
      k_fill<<<1, 2 * kH, 0, stream>>>(stats, 0.f, 2 * kH);
      k_bnstats<<<512, kH, 0, stream>>>(comb, stats, N);
      k_bnapply<<<cdiv(NH, 256), 256, 0, stream>>>(comb, stats, gamma + i * kH,
                                                   beta + i * kH, h, NH, 1.0f / N);
    }
    // gates = emb_t@W_ih.T + hS@W_hh.T + (b_ih+b_hh)
    k_gemm<<<dim3(cdiv(N, 64), (4 * kH) / 64), blk, 0, stream>>>(h, Wip, hS, Whp, biasg,
                                                                 gates, N, kH, 4 * kH);
    k_lstm<<<cdiv(NH, 256), 256, 0, stream>>>(gates, hS, cS, NH);
  }
  k_decoder<<<cdiv((long long)N * kO, 256), 256, 0, stream>>>(hS, W_dec, b_dec,
                                                              (float*)d_out, N);
}
